// DKCA_38371237822853
// MI455X (gfx1250) — matrix operand order repaired and checked by reading, not compiled, not run on hardware
//
#include <hip/hip_runtime.h>

#define B_   8
#define C_   256
#define H_   128
#define W_   128
#define HW_  (H_*W_)
#define D_   16
#define KC_  8
#define KS_  8
#define BN_EPS 1e-5f
#define L2_EPS 1e-12f

typedef __attribute__((ext_vector_type(16))) _Float16 v16h;
typedef __attribute__((ext_vector_type(8)))  float    v8f;

__device__ __forceinline__ float sigmoid_(float x){ return 1.0f/(1.0f+__expf(-x)); }
__device__ __forceinline__ float silu_(float x){ return x*sigmoid_(x); }

// A-fragment element->K mapping for 16-bit WMMA A (16x32), per CDNA5 ISA 7.12.2:
// lane half h: VGPRs 0-3 hold K = 8h..8h+7, VGPRs 4-7 hold K = 16+8h..16+8h+7.
__device__ __host__ __forceinline__ int a_elem_K(int e, int half){
    int j = e >> 1, p = e & 1;
    return (j < 4) ? (8*half + 2*j + p) : (16 + 8*half + 2*(j-4) + p);
}

// ---------------- Kernel 1: per-(b,c) spatial mean / max ----------------
__global__ __launch_bounds__(256)
void gap_gmp_kernel(const float* __restrict__ x, float* __restrict__ gap,
                    float* __restrict__ gmp){
    __shared__ float ss[256], sm[256];
    int bc = blockIdx.x;
    const float4* row = reinterpret_cast<const float4*>(x + (size_t)bc*HW_);
    float s = 0.0f, m = -3.4e38f;
    for (int i = threadIdx.x; i < HW_/4; i += 256){
        float4 v = row[i];
        s += v.x + v.y + v.z + v.w;
        m = fmaxf(m, fmaxf(fmaxf(v.x, v.y), fmaxf(v.z, v.w)));
    }
    ss[threadIdx.x] = s; sm[threadIdx.x] = m;
    __syncthreads();
    for (int off = 128; off > 0; off >>= 1){
        if (threadIdx.x < off){
            ss[threadIdx.x] += ss[threadIdx.x+off];
            sm[threadIdx.x]  = fmaxf(sm[threadIdx.x], sm[threadIdx.x+off]);
        }
        __syncthreads();
    }
    if (threadIdx.x == 0){
        gap[bc] = ss[0] * (1.0f/HW_);
        gmp[bc] = sm[0];
    }
}

// ---------------- Kernel 2: gate_c + BN folding + f16 A-fragment prep ----------------
struct PrepParams {
    const float *gap, *gmp;
    const float *ch_q_w, *ch_q_g, *ch_q_b, *ch_q_m, *ch_q_v;
    const float *ch_proto, *ch_mlp_w, *ch_mlp_b;
    const float *sp_q_w, *sp_q_g, *sp_q_b, *sp_q_m, *sp_q_v;
    const float *sp_k, *sp_v, *sp_gate_w;
    const float *out_w, *out_g, *out_b, *out_m, *out_v;
    float *gate_c, *sp_shift, *ksn, *vg, *out_shift;
    _Float16 *spA, *outA;
};

__global__ __launch_bounds__(256)
void prep_kernel(PrepParams p){
    __shared__ float qc[B_*D_], zc[B_*D_];
    __shared__ float oscale[C_], ssc[D_];
    __shared__ float knorm[KC_], sknorm[KS_];
    int t = threadIdx.x;

    // ---- phase 1: independent folds ----
    if (t < B_*D_){
        int b = t / D_, d = t % D_;
        float acc = 0.0f;
        for (int i = 0; i < C_; ++i) acc += p.gap[b*C_+i] * p.ch_q_w[d*2*C_ + i];
        for (int i = 0; i < C_; ++i) acc += p.gmp[b*C_+i] * p.ch_q_w[d*2*C_ + C_ + i];
        float sc = p.ch_q_g[d] * rsqrtf(p.ch_q_v[d] + BN_EPS);
        float v  = (acc - p.ch_q_m[d]) * sc + p.ch_q_b[d];
        qc[t] = silu_(v);
    }
    if (t < KC_){
        float n = 0.0f;
        for (int d = 0; d < D_; ++d){ float w = p.ch_proto[t*D_+d]; n += w*w; }
        knorm[t] = fmaxf(sqrtf(n), L2_EPS);
    }
    if (t < KS_){
        float n = 0.0f, g = 0.0f;
        for (int d = 0; d < D_; ++d){
            float w = p.sp_k[t*D_+d]; n += w*w;
            g += p.sp_v[t*D_+d] * p.sp_gate_w[d];
        }
        sknorm[t] = fmaxf(sqrtf(n), L2_EPS);
        p.vg[t] = g;
    }
    if (t < C_){
        float sc = p.out_g[t] * rsqrtf(p.out_v[t] + BN_EPS);
        oscale[t] = sc;
        p.out_shift[t] = p.out_b[t] - p.out_m[t]*sc;
    }
    if (t < D_){
        float sc = p.sp_q_g[t] * rsqrtf(p.sp_q_v[t] + BN_EPS);
        ssc[t] = sc;
        p.sp_shift[t] = p.sp_q_b[t] - p.sp_q_m[t]*sc;
    }
    __syncthreads();

    // ---- phase 2: channel prototype attention (tiny) + normalized sp_k ----
    if (t < B_){
        int b = t;
        float qn2 = 0.0f;
        for (int d = 0; d < D_; ++d) qn2 += qc[b*D_+d]*qc[b*D_+d];
        float qinv = 1.0f / fmaxf(sqrtf(qn2), L2_EPS);
        float lg[KC_]; float mx = -3.4e38f;
        for (int k = 0; k < KC_; ++k){
            float s = 0.0f;
            for (int d = 0; d < D_; ++d)
                s += qc[b*D_+d]*qinv * p.ch_proto[k*D_+d] / knorm[k];
            lg[k] = s; mx = fmaxf(mx, s);
        }
        float se = 0.0f;
        for (int k = 0; k < KC_; ++k){ lg[k] = __expf(lg[k]-mx); se += lg[k]; }
        for (int d = 0; d < D_; ++d){
            float z = 0.0f;
            for (int k = 0; k < KC_; ++k) z += (lg[k]/se) * p.ch_proto[k*D_+d];
            zc[b*D_+d] = z;
        }
    }
    if (t < KS_*D_) p.ksn[t] = p.sp_k[t] / sknorm[t / D_];
    __syncthreads();

    // ---- phase 3: gate_c[b][c] ----
    for (int b = 0; b < B_; ++b){
        int c = t;
        float a = p.ch_mlp_b[c];
        for (int d = 0; d < D_; ++d) a += qc[b*D_+d] * p.ch_mlp_w[c*2*D_ + d];
        for (int d = 0; d < D_; ++d) a += zc[b*D_+d] * p.ch_mlp_w[c*2*D_ + D_ + d];
        p.gate_c[b*C_+c] = sigmoid_(a);
    }

    // ---- phase 4: f16 WMMA A-fragments (BN-scale folded into weights) ----
    {   // spA: [kc=8][lane=32][e=16], value = sp_q_w[M][kc*32+K]*ssc[M]
        int kc = t >> 5, L = t & 31;
        int M = L & 15, half = L >> 4;
        float sc = ssc[M];
        _Float16* dst = p.spA + (size_t)t*16;
        #pragma unroll
        for (int e = 0; e < 16; ++e){
            int K = a_elem_K(e, half);
            dst[e] = (_Float16)(p.sp_q_w[(size_t)M*C_ + kc*32 + K] * sc);
        }
    }
    for (int r = 0; r < 16; ++r){   // outA: [mt=16][kc=8][lane=32][e=16]
        int f  = t + r*256;
        int mt = f >> 8;
        int kc = (f >> 5) & 7;
        int L  = f & 31;
        int M = L & 15, half = L >> 4;
        int row = mt*16 + M;
        float sc = oscale[row];
        _Float16* dst = p.outA + (size_t)f*16;
        #pragma unroll
        for (int e = 0; e < 16; ++e){
            int K = a_elem_K(e, half);
            dst[e] = (_Float16)(p.out_w[(size_t)row*C_ + kc*32 + K] * sc);
        }
    }
}

// ---------------- Kernel 3: spatial query GEMM (WMMA) + per-pixel gate_s ----------------
__global__ __launch_bounds__(256)
void gate_s_kernel(const float* __restrict__ x, const float* __restrict__ gate_c,
                   const float* __restrict__ sp_shift, const float* __restrict__ ksn,
                   const float* __restrict__ vg, const float* __restrict__ sp_gate_b,
                   const _Float16* __restrict__ spA, float* __restrict__ gate_s){
    int wave = threadIdx.x >> 5;
    int lane = threadIdx.x & 31;
    long n0 = (long)blockIdx.x*128 + wave*16;   // 16 contiguous pixels per wave
    int b  = (int)(n0 / HW_);
    int p0 = (int)(n0 % HW_);

    const v16h* spAf = reinterpret_cast<const v16h*>(spA);
    v8f acc = {};
    #pragma unroll
    for (int kc = 0; kc < 8; ++kc){
        v16h a = spAf[kc*32 + lane];                  // A fragment (prepacked)
        int bn = lane & 15, bh = lane >> 4;
        v16h bb;
        #pragma unroll
        for (int e = 0; e < 16; ++e){
            int ch = kc*32 + a_elem_K(e, bh);
            float g = gate_c[b*C_ + ch];
            bb[e] = (_Float16)(x[((size_t)(b*C_ + ch))*HW_ + p0 + bn] * g);
        }
        acc = __builtin_amdgcn_wmma_f32_16x16x32_f16(false, a, false, bb,
                                                     (short)0, acc, false, false);
    }

    // Epilogue: lane holds q_d for d = j + 8*h of pixel p0 + (lane&15)
    int h = lane >> 4, col = lane & 15;
    float q[8]; float nsq = 0.0f;
    #pragma unroll
    for (int j = 0; j < 8; ++j){
        float v = silu_(acc[j] + sp_shift[j + 8*h]);
        q[j] = v; nsq += v*v;
    }
    nsq += __shfl_xor(nsq, 16);                       // join both halves
    float qinv = 1.0f / fmaxf(sqrtf(nsq), L2_EPS);
    float lg[KS_]; float mx = -3.4e38f;
    #pragma unroll
    for (int k = 0; k < KS_; ++k){
        float s = 0.0f;
        #pragma unroll
        for (int j = 0; j < 8; ++j) s += q[j]*qinv * ksn[k*D_ + j + 8*h];
        s += __shfl_xor(s, 16);
        lg[k] = s; mx = fmaxf(mx, s);
    }
    float num = 0.0f, den = 0.0f;
    #pragma unroll
    for (int k = 0; k < KS_; ++k){
        float e = __expf(lg[k] - mx);
        den += e; num += e * vg[k];
    }
    float gate = sigmoid_(num/den + sp_gate_b[0]);
    if (h == 0) gate_s[(size_t)b*HW_ + p0 + col] = gate;
}

// ---------------- Kernel 4: output projection GEMM (WMMA) + BN + SiLU ----------------
// Block: 8 waves over 64 pixels; wave = (mg: half of Cout) x (ng: 16-pixel group).
// One B-fragment build feeds 8 WMMAs (amortizes f32->f16 convert VALU 8x).
__global__ __launch_bounds__(256)
void out_gemm_kernel(const float* __restrict__ x, const float* __restrict__ gate_c,
                     const float* __restrict__ gate_s, const _Float16* __restrict__ outA,
                     const float* __restrict__ out_shift, float* __restrict__ out){
    __shared__ __align__(16) float gc[C_];
    __shared__ __align__(16) float osh[C_];
    __shared__ __align__(16) float gs[64];
    int t = threadIdx.x;
    long n0 = (long)blockIdx.x*64;                    // 64 pixels / block
    int b  = (int)(n0 / HW_);
    int pb = (int)(n0 % HW_);

    // Stage per-block tables into LDS via the CDNA5 async global->LDS path
    // (ASYNCcnt-tracked; bulk LDS fill without a VGPR round-trip).
    if (t < 144){
        unsigned ldsa;
        const float* src;
        if (t < 64){        ldsa = (unsigned)(size_t)&gc[0]  + (unsigned)t*16u;
                            src  = gate_c + b*C_ + t*4; }
        else if (t < 128){  ldsa = (unsigned)(size_t)&osh[0] + (unsigned)(t-64)*16u;
                            src  = out_shift + (t-64)*4; }
        else {              ldsa = (unsigned)(size_t)&gs[0]  + (unsigned)(t-128)*16u;
                            src  = gate_s + (size_t)b*HW_ + pb + (t-128)*4; }
        asm volatile("global_load_async_to_lds_b128 %0, %1, off"
                     :: "v"(ldsa), "v"(src) : "memory");
    }
    asm volatile("s_wait_asynccnt 0x0" ::: "memory");
    __syncthreads();

    int wave = t >> 5, lane = t & 31;
    int mg = wave & 1;                                // 8 M-tiles per wave
    int ng = wave >> 1;                               // 4 pixel groups of 16
    int p0 = pb + ng*16;

    const v16h* oAf = reinterpret_cast<const v16h*>(outA);
    v8f acc[8] = {};
    #pragma unroll
    for (int kc = 0; kc < 8; ++kc){
        int ch = kc*32 + lane;                        // prefetch row: lane = in-channel
        const float* xr = x + ((size_t)(b*C_ + ch))*HW_ + p0;
        if (kc < 7) __builtin_prefetch(xr + 32*(size_t)HW_, 0, 1);
        int bn = lane & 15, bh = lane >> 4;
        float gsv = gs[ng*16 + bn];
        v16h bb;
        #pragma unroll
        for (int e = 0; e < 16; ++e){
            int chk = kc*32 + a_elem_K(e, bh);
            float gcv = gc[chk];
            // (x + x_c + x_s) = x * (1 + gate_c*(1 + gate_s)) = x * fma(gc, 1+gs, 1)
            bb[e] = (_Float16)(x[((size_t)(b*C_ + chk))*HW_ + p0 + bn] * fmaf(gcv, 1.0f + gsv, 1.0f));
        }
        #pragma unroll
        for (int i = 0; i < 8; ++i){
            int mt = mg*8 + i;
            v16h a = oAf[((size_t)mt*8 + kc)*32 + lane];
            acc[i] = __builtin_amdgcn_wmma_f32_16x16x32_f16(false, a, false, bb,
                                                            (short)0, acc[i], false, false);
        }
    }

    int h = lane >> 4, col = lane & 15;
    #pragma unroll
    for (int i = 0; i < 8; ++i){
        int mt = mg*8 + i;
        #pragma unroll
        for (int j = 0; j < 8; ++j){
            int o = mt*16 + j + 8*h;                  // out channel (BN scale prefolded)
            float y = acc[i][j] + osh[o];
            // out is write-once; NT store keeps x resident in the 192MB L2
            __builtin_nontemporal_store(silu_(y),
                &out[((size_t)(b*C_ + o))*HW_ + p0 + col]);
        }
    }
}

// ---------------- host launcher ----------------
extern "C" void kernel_launch(void* const* d_in, const int* in_sizes, int n_in,
                              void* d_out, int out_size, void* d_ws, size_t ws_size,
                              hipStream_t stream){
    const float* x         = (const float*)d_in[0];
    const float* ch_q_w    = (const float*)d_in[1];
    const float* ch_q_g    = (const float*)d_in[2];
    const float* ch_q_b    = (const float*)d_in[3];
    const float* ch_q_m    = (const float*)d_in[4];
    const float* ch_q_v    = (const float*)d_in[5];
    const float* ch_proto  = (const float*)d_in[6];
    const float* ch_mlp_w  = (const float*)d_in[7];
    const float* ch_mlp_b  = (const float*)d_in[8];
    const float* sp_q_w    = (const float*)d_in[9];
    const float* sp_q_g    = (const float*)d_in[10];
    const float* sp_q_b    = (const float*)d_in[11];
    const float* sp_q_m    = (const float*)d_in[12];
    const float* sp_q_v    = (const float*)d_in[13];
    const float* sp_k      = (const float*)d_in[14];
    const float* sp_v      = (const float*)d_in[15];
    const float* sp_gate_w = (const float*)d_in[16];
    const float* sp_gate_b = (const float*)d_in[17];
    const float* out_w     = (const float*)d_in[18];
    const float* out_g     = (const float*)d_in[19];
    const float* out_b     = (const float*)d_in[20];
    const float* out_m     = (const float*)d_in[21];
    const float* out_v     = (const float*)d_in[22];

    // workspace layout (floats first, then 64B-aligned f16 fragments)
    float* ws_f      = (float*)d_ws;
    float* gap       = ws_f;                 // 2048
    float* gmp       = ws_f + 2048;          // 2048
    float* gate_c    = ws_f + 4096;          // 2048
    float* gate_s    = ws_f + 6144;          // 131072
    float* sp_shift  = ws_f + 137216;        // 16
    float* ksn       = ws_f + 137232;        // 128
    float* vg        = ws_f + 137360;        // 8
    float* out_shift = ws_f + 137368;        // 256 -> ends 137624, pad to 137632
    _Float16* spA    = (_Float16*)(ws_f + 137632); // 8*32*16   = 4096 halves
    _Float16* outA   = spA + 4096;                 // 16*8*32*16 = 65536 halves

    gap_gmp_kernel<<<B_*C_, 256, 0, stream>>>(x, gap, gmp);

    PrepParams P;
    P.gap = gap; P.gmp = gmp;
    P.ch_q_w = ch_q_w; P.ch_q_g = ch_q_g; P.ch_q_b = ch_q_b; P.ch_q_m = ch_q_m; P.ch_q_v = ch_q_v;
    P.ch_proto = ch_proto; P.ch_mlp_w = ch_mlp_w; P.ch_mlp_b = ch_mlp_b;
    P.sp_q_w = sp_q_w; P.sp_q_g = sp_q_g; P.sp_q_b = sp_q_b; P.sp_q_m = sp_q_m; P.sp_q_v = sp_q_v;
    P.sp_k = sp_k; P.sp_v = sp_v; P.sp_gate_w = sp_gate_w;
    P.out_w = out_w; P.out_g = out_g; P.out_b = out_b; P.out_m = out_m; P.out_v = out_v;
    P.gate_c = gate_c; P.sp_shift = sp_shift; P.ksn = ksn; P.vg = vg; P.out_shift = out_shift;
    P.spA = spA; P.outA = outA;
    prep_kernel<<<1, 256, 0, stream>>>(P);

    gate_s_kernel<<<(B_*HW_)/128, 256, 0, stream>>>(x, gate_c, sp_shift, ksn, vg,
                                                    sp_gate_b, spA, gate_s);
    out_gemm_kernel<<<(B_*HW_)/64, 256, 0, stream>>>(x, gate_c, gate_s, outA,
                                                     out_shift, (float*)d_out);
}
